// ContrastiveLoss_34720515621658
// MI455X (gfx1250) — matrix operand order repaired and checked by reading, not compiled, not run on hardware
//
#include <hip/hip_runtime.h>
#include <math.h>

typedef __attribute__((ext_vector_type(16))) _Float16 v16h;
typedef __attribute__((ext_vector_type(8)))  _Float16 v8h;
typedef __attribute__((ext_vector_type(8)))  float    v8f;

union Frag16 { v16h v; v8h h[2]; };

#define INV_T   100.0f
#define NEG_INF (-__builtin_inff())

// Convert a generic pointer to a __shared__ object into its 32-bit LDS byte offset.
__device__ __forceinline__ unsigned lds_off_u32(const void* p) {
  return (unsigned)(unsigned long long)(__attribute__((address_space(3))) const void*)p;
}

// Async global->LDS copy of 32 bytes (two b128 beats), tracked by ASYNCcnt.
// IOFFSET applies to both the LDS and global addresses (ISA 15.18.3).
__device__ __forceinline__ void async_copy_b256(unsigned ldsOff, const _Float16* gp) {
  asm volatile("global_load_async_to_lds_b128 %0, %1, off"
               :: "v"(ldsOff), "v"(gp) : "memory");
  asm volatile("global_load_async_to_lds_b128 %0, %1, off offset:16"
               :: "v"(ldsOff), "v"(gp) : "memory");
}

// ---------- row L2 normalize + f32 -> f16 (block = 256 threads, one row, D=1024) ----------
__global__ void nrm_f16_kernel(const float* __restrict__ src,
                               _Float16* __restrict__ dst, int D) {
  const int row = blockIdx.x;
  const int t = threadIdx.x;
  const float* p = src + (size_t)row * D;
  float x0 = p[t], x1 = p[t + 256], x2 = p[t + 512], x3 = p[t + 768];
  float ss = x0 * x0 + x1 * x1 + x2 * x2 + x3 * x3;
#pragma unroll
  for (int m = 16; m >= 1; m >>= 1) ss += __shfl_xor(ss, m, 32);
  __shared__ float red[8];
  __shared__ float s_scale;
  if ((t & 31) == 0) red[t >> 5] = ss;
  __syncthreads();
  if (t == 0) {
    float tot = 0.f;
#pragma unroll
    for (int i = 0; i < 8; ++i) tot += red[i];
    s_scale = 1.0f / fmaxf(sqrtf(tot), 1e-12f);
  }
  __syncthreads();
  const float sc = s_scale;
  _Float16* q = dst + (size_t)row * D;
  q[t]       = (_Float16)(x0 * sc);
  q[t + 256] = (_Float16)(x1 * sc);
  q[t + 512] = (_Float16)(x2 * sc);
  q[t + 768] = (_Float16)(x3 * sc);
}

// ---------- diag_i = dot(xn_i, yn_i) / T  (one wave per row) ----------
__global__ void diag_kernel(const _Float16* __restrict__ Xn,
                            const _Float16* __restrict__ Yn,
                            float* __restrict__ diag, int D) {
  const int lane = threadIdx.x & 31;
  const int w = threadIdx.x >> 5;
  const int row = blockIdx.x * 8 + w;
  const _Float16* xp = Xn + (size_t)row * D;
  const _Float16* yp = Yn + (size_t)row * D;
  float acc = 0.f;
  for (int d = lane; d < D; d += 32) acc += (float)xp[d] * (float)yp[d];
#pragma unroll
  for (int m = 16; m >= 1; m >>= 1) acc += __shfl_xor(acc, m, 32);
  if (lane == 0) diag[row] = acc * INV_T;
}

// ---------- fused WMMA GEMM + streaming row-logsumexp partials ----------
// Block: 256 threads (8 waves). Block tile: 128 rows x (tilesPerChunk*128) cols.
// B tiles (128 cols x 32 K) triple-buffered in LDS, filled by async global->LDS
// copies pipelined one K-step ahead; one barrier per K-step.
#define BSTRIDE 40   // padded row stride in halves (bank-conflict-free ds_load_b128)
#define BBUF    (128 * BSTRIDE)

__global__ void __launch_bounds__(256)
gemm_lse_kernel(const _Float16* __restrict__ A,   // [N][D] normalized, row-major
                const _Float16* __restrict__ B,   // [N][D] normalized, row-major
                float* __restrict__ pMax,         // [chunks][N]
                float* __restrict__ pSum,         // [chunks][N]
                int N, int D, int tilesPerChunk) {
  __shared__ alignas(16) _Float16 Bs[3 * BBUF];

  const int lane = threadIdx.x & 31;
  const int wave = threadIdx.x >> 5;
  const int n4   = lane & 15;   // row-in-tile for A frag / col-in-tile for B,C frags
  const int hi   = lane >> 4;   // half-group: K sub-range (A,B) / M sub-range (C)
  const int chunk = blockIdx.x;
  const int rowBase = blockIdx.y * 128 + wave * 16;
  const int colChunkBase = chunk * (tilesPerChunk * 128);

  const _Float16* arow = A + (size_t)(rowBase + n4) * D;

  float runM[8], runS[8];
#pragma unroll
  for (int r = 0; r < 8; ++r) { runM[r] = NEG_INF; runS[r] = 0.f; }

  const int cStage = threadIdx.x >> 1;  // 0..127 (column staged by this thread)
  const int hStage = threadIdx.x & 1;   // which 16-half K chunk
  const unsigned ldsStageBase =
      lds_off_u32(&Bs[0]) + (unsigned)(cStage * BSTRIDE + hStage * 16) * 2u;

  for (int ct = 0; ct < tilesPerChunk; ++ct) {
    const int colB = colChunkBase + ct * 128;
    const _Float16* brow = B + (size_t)(colB + cStage) * D + hStage * 16;

    v8f zero = {};
    v8f acc[8];
#pragma unroll
    for (int j = 0; j < 8; ++j) acc[j] = zero;

    // prologue: stage K-step 0 into buffer 0
    async_copy_b256(ldsStageBase, brow);

    int buf = 0;
    for (int k0 = 0; k0 < D; k0 += 32) {
      const bool more = (k0 + 32) < D;
      const int nextBuf = (buf == 2) ? 0 : buf + 1;
      if (more) {
        async_copy_b256(ldsStageBase + (unsigned)(nextBuf * BBUF) * 2u,
                        brow + (k0 + 32));
      }
      if (more) {
        asm volatile("s_wait_asynccnt 0x2" ::: "memory");  // older copy retired
      } else {
        asm volatile("s_wait_asynccnt 0x0" ::: "memory");
      }
      __syncthreads();  // all waves' copies for this K-step visible

      // A fragment (16x32 f16): lane m=n4, half-group hi -> K = {hi*16..hi*16+15}
      Frag16 af;
      af.h[0] = *(const v8h*)(arow + k0 + hi * 16);
      af.h[1] = *(const v8h*)(arow + k0 + hi * 16 + 8);

      // preload all 8 B fragments, then issue 8 back-to-back WMMAs
      const _Float16* bbase = &Bs[buf * BBUF + n4 * BSTRIDE + hi * 16];
      Frag16 bf[8];
#pragma unroll
      for (int j = 0; j < 8; ++j) {
        const _Float16* s = bbase + j * (16 * BSTRIDE);
        bf[j].h[0] = *(const v8h*)(s);
        bf[j].h[1] = *(const v8h*)(s + 8);
      }
#pragma unroll
      for (int j = 0; j < 8; ++j) {
        acc[j] = __builtin_amdgcn_wmma_f32_16x16x32_f16(
            false, af.v, false, bf[j].v, (short)0, acc[j], false, false);
      }
      buf = nextBuf;
    }

    // Online LSE update for this 16-row x 128-col tile.
    // C layout: lane holds col n4, rows r + 8*hi in VGPR r.
#pragma unroll
    for (int r = 0; r < 8; ++r) {
      float v = NEG_INF;
#pragma unroll
      for (int j = 0; j < 8; ++j) v = fmaxf(v, acc[j][r] * INV_T);
#pragma unroll
      for (int m = 8; m >= 1; m >>= 1) v = fmaxf(v, __shfl_xor(v, m, 32));
      float s = 0.f;
#pragma unroll
      for (int j = 0; j < 8; ++j) s += __expf(acc[j][r] * INV_T - v);
#pragma unroll
      for (int m = 8; m >= 1; m >>= 1) s += __shfl_xor(s, m, 32);
      const float nM = fmaxf(runM[r], v);
      runS[r] = runS[r] * __expf(runM[r] - nM) + s * __expf(v - nM);
      runM[r] = nM;
    }
  }

  if (n4 == 0) {
#pragma unroll
    for (int r = 0; r < 8; ++r) {
      const int row = rowBase + hi * 8 + r;
      pMax[(size_t)chunk * N + row] = runM[r];
      pSum[(size_t)chunk * N + row] = runS[r];
    }
  }
}

// ---------- combine per-chunk (max,sum) partials into lse per row ----------
__global__ void lse_combine_kernel(const float* __restrict__ pMax,
                                   const float* __restrict__ pSum,
                                   float* __restrict__ lse, int N, int chunks) {
  const int i = blockIdx.x * blockDim.x + threadIdx.x;
  if (i >= N) return;
  float M = NEG_INF;
  for (int c = 0; c < chunks; ++c) M = fmaxf(M, pMax[(size_t)c * N + i]);
  float S = 0.f;
  for (int c = 0; c < chunks; ++c)
    S += pSum[(size_t)c * N + i] * __expf(pMax[(size_t)c * N + i] - M);
  lse[i] = M + __logf(S);
}

// ---------- final scalar loss ----------
__global__ void final_loss_kernel(const float* __restrict__ rowLse,
                                  const float* __restrict__ colLse,
                                  const float* __restrict__ diag,
                                  float* __restrict__ out, int N) {
  __shared__ float red[256];
  float s = 0.f;
  for (int i = threadIdx.x; i < N; i += 256)
    s += rowLse[i] + colLse[i] - 2.0f * diag[i];
  red[threadIdx.x] = s;
  __syncthreads();
  for (int stride = 128; stride > 0; stride >>= 1) {
    if (threadIdx.x < stride) red[threadIdx.x] += red[threadIdx.x + stride];
    __syncthreads();
  }
  if (threadIdx.x == 0) out[0] = 0.5f * red[0] / (float)N;
}

extern "C" void kernel_launch(void* const* d_in, const int* in_sizes, int n_in,
                              void* d_out, int out_size, void* d_ws, size_t ws_size,
                              hipStream_t stream) {
  (void)n_in; (void)out_size; (void)ws_size;
  const int D = 1024;
  const int N = in_sizes[0] / D;                  // 8192
  const int CHUNKS = 8;
  const int tilesPerChunk = (N / CHUNKS) / 128;   // 8

  const float* x = (const float*)d_in[0];
  const float* y = (const float*)d_in[1];
  float* out = (float*)d_out;

  char* w = (char*)d_ws;
  size_t off = 0;
  auto carve = [&](size_t bytes) -> char* {
    char* p = w + off;
    off += (bytes + 255) & ~(size_t)255;
    return p;
  };
  _Float16* Xn = (_Float16*)carve((size_t)N * D * sizeof(_Float16));
  _Float16* Yn = (_Float16*)carve((size_t)N * D * sizeof(_Float16));
  float* diag   = (float*)carve((size_t)N * sizeof(float));
  float* pRM    = (float*)carve((size_t)CHUNKS * N * sizeof(float));
  float* pRS    = (float*)carve((size_t)CHUNKS * N * sizeof(float));
  float* pCM    = (float*)carve((size_t)CHUNKS * N * sizeof(float));
  float* pCS    = (float*)carve((size_t)CHUNKS * N * sizeof(float));
  float* rowLse = (float*)carve((size_t)N * sizeof(float));
  float* colLse = (float*)carve((size_t)N * sizeof(float));

  nrm_f16_kernel<<<N, 256, 0, stream>>>(x, Xn, D);
  nrm_f16_kernel<<<N, 256, 0, stream>>>(y, Yn, D);
  diag_kernel<<<N / 8, 256, 0, stream>>>(Xn, Yn, diag, D);

  dim3 grid(CHUNKS, N / 128);
  // row-wise LSE of Xn*Yn^T
  gemm_lse_kernel<<<grid, 256, 0, stream>>>(Xn, Yn, pRM, pRS, N, D, tilesPerChunk);
  // column-wise LSE == row-wise LSE of Yn*Xn^T
  gemm_lse_kernel<<<grid, 256, 0, stream>>>(Yn, Xn, pCM, pCS, N, D, tilesPerChunk);

  lse_combine_kernel<<<(N + 255) / 256, 256, 0, stream>>>(pRM, pRS, rowLse, N, CHUNKS);
  lse_combine_kernel<<<(N + 255) / 256, 256, 0, stream>>>(pCM, pCS, colLse, N, CHUNKS);
  final_loss_kernel<<<1, 256, 0, stream>>>(rowLse, colLse, diag, out, N);
}
